// HomoDecoder_72146860638640
// MI455X (gfx1250) — matrix operand order repaired and checked by reading, not compiled, not run on hardware
//
#include <hip/hip_runtime.h>
#include <stdint.h>

typedef __attribute__((ext_vector_type(16))) _Float16 v16h;
typedef __attribute__((ext_vector_type(8)))  _Float16 v8h;
typedef __attribute__((ext_vector_type(8)))  float    v8f;

#define HID 128
#define LN_EPS 1e-5f
#define HSTR 136              // padded f16 row stride (272 B) for per-edge h rows

// LDS layout (bytes)
#define PAR_OFF    163840     // after 81920 f16 of prepacked A(W^T) fragments
#define H_OFF      168960     // after 10 x 128 f32 params
#define H_WAVE     (32*HSTR)  // 32 edge-rows per wave
#define SMEM_BYTES (H_OFF + 8*H_WAVE*2)   // 238592

// cross-half (lane ^ 16) exchange-add via ds_swizzle SWAPX16 (xor=0x10, and=0x1f)
#define SWZ_ADD(v, IMM) ((v) + __int_as_float(__builtin_amdgcn_ds_swizzle(__float_as_int(v), (IMM))))

// ---------------------------------------------------------------------------
// Prepack W0/W1/W2 (f32 row-major [K][128]) into f16 WMMA *A-fragment* order
// for the transposed GEMM  Y^T[128][16] = W^T[128][K] x H^T[K][16]:
//   elem idx = ((ktG*8 + mt)*32 + lane)*16 + j
//   m = mt*16 + (lane&15) ; kb = (lane<16 ? 0 : 16)
//   k = ktG_local*32 + kb + j                          (contiguous K order)
//   value = W[k][m]   (i.e. W^T[m][k])
// ktG frags: 0..11 -> W0, 12..15 -> W1, 16..19 -> W2.
// ---------------------------------------------------------------------------
__global__ __launch_bounds__(256) void prepack_kernel(
    const float* __restrict__ W0, const float* __restrict__ W1,
    const float* __restrict__ W2, _Float16* __restrict__ dst)
{
  int idx = blockIdx.x * 256 + threadIdx.x;
  if (idx >= 20 * 8 * 32 * 16) return;
  int j    = idx & 15;
  int lane = (idx >> 4) & 31;
  int mt   = (idx >> 9) & 7;
  int frag = idx >> 12;
  const float* W; int kt;
  if (frag < 12)      { W = W0; kt = frag; }
  else if (frag < 16) { W = W1; kt = frag - 12; }
  else                { W = W2; kt = frag - 16; }
  int kb = (lane < 16) ? 0 : 16;
  int k  = kt * 32 + kb + j;
  int m  = mt * 16 + (lane & 15);
  dst[idx] = (_Float16)W[k * HID + m];
}

// ---------------------------------------------------------------------------
__device__ __forceinline__ float hsum8(v8f v)
{
  float a = (v[0] + v[4]) + (v[1] + v[5]);
  float b = (v[2] + v[6]) + (v[3] + v[7]);
  return a + b;
}

// B operand (H^T fragment): 16 consecutive f16 of one edge-row from LDS.
__device__ __forceinline__ v16h load_b_lds(const _Float16* __restrict__ p)
{
  v8h lo = *(const v8h*)p;
  v8h hi = *(const v8h*)(p + 8);
  v16h b;
#pragma unroll
  for (int i = 0; i < 8; ++i) { b[i] = lo[i]; b[8 + i] = hi[i]; }
  return b;
}

// B operand built from 16 consecutive gathered f32 features in global memory.
__device__ __forceinline__ v16h load_b_gmem(const float* __restrict__ p)
{
  float4 f0 = ((const float4*)p)[0];
  float4 f1 = ((const float4*)p)[1];
  float4 f2 = ((const float4*)p)[2];
  float4 f3 = ((const float4*)p)[3];
  v16h b;
  b[0]  = (_Float16)f0.x; b[1]  = (_Float16)f0.y; b[2]  = (_Float16)f0.z; b[3]  = (_Float16)f0.w;
  b[4]  = (_Float16)f1.x; b[5]  = (_Float16)f1.y; b[6]  = (_Float16)f1.z; b[7]  = (_Float16)f1.w;
  b[8]  = (_Float16)f2.x; b[9]  = (_Float16)f2.y; b[10] = (_Float16)f2.z; b[11] = (_Float16)f2.w;
  b[12] = (_Float16)f3.x; b[13] = (_Float16)f3.y; b[14] = (_Float16)f3.z; b[15] = (_Float16)f3.w;
  return b;
}

// One K-tile: 8 weight A-fragments, each reused for BOTH edge-tiles.
// 2-deep software pipeline on the A loads so each ds_load is issued ~4 WMMAs
// ahead of its consumer (kills the exposed s_wait_dscnt latency).
__device__ __forceinline__ void gemm_step(const _Float16* __restrict__ aBase,
                                          v16h b0, v16h b1,
                                          v8f a0[8], v8f a1[8], int lane)
{
  const _Float16* ap = aBase + (lane << 4);
  v16h aCur  = *(const v16h*)ap;
  v16h aNext = *(const v16h*)(ap + 512);
#pragma unroll
  for (int mt = 0; mt < 8; ++mt) {
    v16h aNN;
    if (mt < 6) aNN = *(const v16h*)(ap + (mt + 2) * 512);
    a0[mt] = __builtin_amdgcn_wmma_f32_16x16x32_f16(
        false, aCur, false, b0, (short)0, a0[mt], false, false);
    a1[mt] = __builtin_amdgcn_wmma_f32_16x16x32_f16(
        false, aCur, false, b1, (short)0, a1[mt], false, false);
    aCur = aNext;
    if (mt < 6) aNext = aNN;
  }
}

// LayerNorm(+bias)+ReLU. C layout: per lane, acc[mt][r] = feature
// (mt*16 + 8*half + r) of ONE edge (= lane&15 of its tile). So the 128-wide
// reduction is 8 vector adds + an in-register horizontal sum + one SWAPX16.
__device__ __forceinline__ void ln_relu2(v8f acc[8], const float* __restrict__ par, int half)
{
  const float* bp = par;
  const float* gp = par + 128;
  const float* ep = par + 256;
  const int fo = 8 * half;
  v8f vs = {}; v8f vq = {};
#pragma unroll
  for (int mt = 0; mt < 8; ++mt) {
    v8f bv = *(const v8f*)(bp + mt * 16 + fo);
    acc[mt] += bv;
    vs += acc[mt];
    vq += acc[mt] * acc[mt];
  }
  float hs = hsum8(vs);
  float hq = hsum8(vq);
  hs = SWZ_ADD(hs, 0x401F);     // lane ^ 16 : combine the two feature halves
  hq = SWZ_ADD(hq, 0x401F);
  float mean = hs * (1.0f / HID);
  float var  = hq * (1.0f / HID) - mean * mean;
  float rst  = rsqrtf(var + LN_EPS);
#pragma unroll
  for (int mt = 0; mt < 8; ++mt) {
    v8f gv = *(const v8f*)(gp + mt * 16 + fo);
    v8f ev = *(const v8f*)(ep + mt * 16 + fo);
    v8f y = (acc[mt] - mean) * rst * gv + ev;
#pragma unroll
    for (int r = 0; r < 8; ++r) y[r] = y[r] > 0.f ? y[r] : 0.f;
    acc[mt] = y;
  }
}

// Write one edge-row back: 8 contiguous f16 per M-tile -> one 16B LDS store each.
__device__ __forceinline__ void store_h2(const v8f acc[8], _Float16* __restrict__ hrow, int half)
{
#pragma unroll
  for (int mt = 0; mt < 8; ++mt) {
    v8h hv;
#pragma unroll
    for (int r = 0; r < 8; ++r) hv[r] = (_Float16)acc[mt][r];
    *(v8h*)(hrow + mt * 16 + 8 * half) = hv;
  }
}

// ---------------------------------------------------------------------------
// Main kernel: 256 threads = 8 waves; each wave handles 32 edges (2 N-tiles).
// ---------------------------------------------------------------------------
__global__ __launch_bounds__(256) void edge_mlp_kernel(
    const float* __restrict__ x, const long long* __restrict__ ei,
    const float* __restrict__ efeat,
    const float* __restrict__ b0, const float* __restrict__ g0, const float* __restrict__ be0,
    const float* __restrict__ b1, const float* __restrict__ g1, const float* __restrict__ be1,
    const float* __restrict__ b2, const float* __restrict__ g2, const float* __restrict__ be2,
    const float* __restrict__ W3, const float* __restrict__ b3,
    const _Float16* __restrict__ wpack, float* __restrict__ out, int E)
{
  extern __shared__ char smem[];
  _Float16* sW   = (_Float16*)smem;
  float*    sPar = (float*)(smem + PAR_OFF);
  _Float16* sH   = (_Float16*)(smem + H_OFF);

  const int tid = threadIdx.x;

  // Stage prepacked weights (163840 B = 10240 uint4, 40 per thread) into LDS.
  {
    const uint4* src4 = (const uint4*)wpack;
    uint4* dst4 = (uint4*)smem;
#pragma unroll
    for (int i = 0; i < 40; ++i) dst4[tid + i * 256] = src4[tid + i * 256];
  }
  if (tid < 128) {
    sPar[0 * 128 + tid] = b0[tid];  sPar[1 * 128 + tid] = g0[tid];  sPar[2 * 128 + tid] = be0[tid];
    sPar[3 * 128 + tid] = b1[tid];  sPar[4 * 128 + tid] = g1[tid];  sPar[5 * 128 + tid] = be1[tid];
    sPar[6 * 128 + tid] = b2[tid];  sPar[7 * 128 + tid] = g2[tid];  sPar[8 * 128 + tid] = be2[tid];
    sPar[9 * 128 + tid] = W3[tid];
  }
  __syncthreads();

  const int wave = tid >> 5, lane = tid & 31;
  const int c = lane & 15, half = lane >> 4;
  const int ko = 16 * half;                       // K-chunk base within a 32-K tile
  const long long tileBase = (long long)blockIdx.x * 256 + wave * 32;

  long long e0 = tileBase + c;      if (e0 > (long long)E - 1) e0 = (long long)E - 1;
  long long e1 = tileBase + 16 + c; if (e1 > (long long)E - 1) e1 = (long long)E - 1;
  const long long sn0 = ei[e0], dn0 = ei[(long long)E + e0];
  const long long sn1 = ei[e1], dn1 = ei[(long long)E + e1];
  const float* r0[3] = { x + sn0 * HID, x + dn0 * HID, efeat + e0 * HID };
  const float* r1[3] = { x + sn1 * HID, x + dn1 * HID, efeat + e1 * HID };

  _Float16* hrow0 = sH + wave * H_WAVE + c * HSTR;   // edge-row of tile 0
  _Float16* hrow1 = hrow0 + 16 * HSTR;               // edge-row of tile 1

  // ---- Layer 0: K = 384, B built from gathered f32 features (16 contiguous) ----
  v8f acc0[8] = {}; v8f acc1[8] = {};
#pragma unroll
  for (int kt = 0; kt < 12; ++kt) {
    const int kg = kt * 32 + ko;                 // 32-K tiles never straddle 128 seams
    v16h bv0 = load_b_gmem(r0[kg >> 7] + (kg & 127));
    v16h bv1 = load_b_gmem(r1[kg >> 7] + (kg & 127));
    gemm_step(sW + kt * 4096, bv0, bv1, acc0, acc1, lane);
  }
  ln_relu2(acc0, sPar + 0, half);
  ln_relu2(acc1, sPar + 0, half);
  store_h2(acc0, hrow0, half);
  store_h2(acc1, hrow1, half);

  // ---- Layer 1 ----
  v8f c0[8] = {}; v8f c1[8] = {};
#pragma unroll
  for (int kt = 0; kt < 4; ++kt) {
    v16h bv0 = load_b_lds(hrow0 + kt * 32 + ko);
    v16h bv1 = load_b_lds(hrow1 + kt * 32 + ko);
    gemm_step(sW + 49152 + kt * 4096, bv0, bv1, c0, c1, lane);
  }
  ln_relu2(c0, sPar + 384, half);
  ln_relu2(c1, sPar + 384, half);
  store_h2(c0, hrow0, half);        // same-wave LDS ordering: prior reads precede writes
  store_h2(c1, hrow1, half);

  // ---- Layer 2 ----
  v8f d0[8] = {}; v8f d1[8] = {};
#pragma unroll
  for (int kt = 0; kt < 4; ++kt) {
    v16h bv0 = load_b_lds(hrow0 + kt * 32 + ko);
    v16h bv1 = load_b_lds(hrow1 + kt * 32 + ko);
    gemm_step(sW + 65536 + kt * 4096, bv0, bv1, d0, d1, lane);
  }
  ln_relu2(d0, sPar + 768, half);
  ln_relu2(d1, sPar + 768, half);

  // ---- Final 128 -> 1 dot: per-lane 64-feature partial + one SWAPX16 ----
  const float* wp = sPar + 9 * 128 + 8 * half;
  v8f vp0 = {}; v8f vp1 = {};
#pragma unroll
  for (int mt = 0; mt < 8; ++mt) {
    v8f wv = *(const v8f*)(wp + mt * 16);
    vp0 += d0[mt] * wv;
    vp1 += d1[mt] * wv;
  }
  float p0 = hsum8(vp0);
  float p1 = hsum8(vp1);
  p0 = SWZ_ADD(p0, 0x401F);
  p1 = SWZ_ADD(p1, 0x401F);

  if (half == 0) {                 // lanes 0..15 store 16 contiguous outputs per tile
    const float bb = b3[0];
    long long o0 = tileBase + c;      if (o0 < (long long)E) out[o0] = p0 + bb;
    long long o1 = tileBase + 16 + c; if (o1 < (long long)E) out[o1] = p1 + bb;
  }
}

// ---------------------------------------------------------------------------
extern "C" void kernel_launch(void* const* d_in, const int* in_sizes, int n_in,
                              void* d_out, int out_size, void* d_ws, size_t ws_size,
                              hipStream_t stream)
{
  (void)in_sizes; (void)n_in; (void)ws_size;
  const float*     x   = (const float*)d_in[0];
  const long long* ei  = (const long long*)d_in[1];   // int64 edge_index [2,E]
  const float*     e   = (const float*)d_in[2];
  const float*     W0  = (const float*)d_in[3];
  const float*     b0  = (const float*)d_in[4];
  const float*     g0  = (const float*)d_in[5];
  const float*     be0 = (const float*)d_in[6];
  const float*     W1  = (const float*)d_in[7];
  const float*     b1  = (const float*)d_in[8];
  const float*     g1  = (const float*)d_in[9];
  const float*     be1 = (const float*)d_in[10];
  const float*     W2  = (const float*)d_in[11];
  const float*     b2  = (const float*)d_in[12];
  const float*     g2  = (const float*)d_in[13];
  const float*     be2 = (const float*)d_in[14];
  const float*     W3  = (const float*)d_in[15];
  const float*     b3  = (const float*)d_in[16];

  _Float16* wpack = (_Float16*)d_ws;      // 163840 bytes of prepacked f16 A-fragments
  const int E = out_size;

  prepack_kernel<<<(20 * 8 * 32 * 16 + 255) / 256, 256, 0, stream>>>(W0, W1, W2, wpack);

  const int blocks = (E + 255) / 256;     // 256 edges per block (8 waves x 32)
  edge_mlp_kernel<<<blocks, 256, SMEM_BYTES, stream>>>(
      x, ei, e, b0, g0, be0, b1, g1, be1, b2, g2, be2, W3, b3,
      wpack, (float*)d_out, E);
}
